// FusedOverEmbedding_49065706390110
// MI455X (gfx1250) — matrix operand order repaired and checked by reading, not compiled, not run on hardware
//
#include <hip/hip_runtime.h>
#include <hip/hip_bf16.h>

// ---------------- problem constants ----------------
#define S_LEN   4096
#define D_DIM   4096
#define OH_DIM  512
#define G_GRAMS 8
#define M_BASE  50021
#define VT_BASE 32000

typedef __attribute__((ext_vector_type(8)))  float          v8f;
typedef __attribute__((ext_vector_type(16))) __bf16         v16bf;
typedef __attribute__((ext_vector_type(8)))  unsigned short us8;

__device__ __forceinline__ unsigned short f2bf(float f) {
    unsigned int u = __builtin_bit_cast(unsigned int, f);
    u += 0x7FFFu + ((u >> 16) & 1u);            // round-to-nearest-even
    return (unsigned short)(u >> 16);
}

// ---------------------------------------------------------------------------
// Kernel 1: hashed n-gram id (exact int64 modular arithmetic) + gather the
// oe_weight row + convert to bf16.  One block per (s, g).
// ---------------------------------------------------------------------------
__global__ __launch_bounds__(256)
void gather_convert_A(const int* __restrict__ input_ids,
                      const float* __restrict__ oe_weight,
                      unsigned short* __restrict__ A)
{
    const int s = blockIdx.x;
    const int g = blockIdx.y;

    long long t0 = input_ids[s];
    long long t1 = (s >= 1) ? input_ids[s - 1] : 0;
    long long t2 = (s >= 2) ? input_ids[s - 2] : 0;

    const long long mod = (long long)M_BASE + 2 * g + 1;
    const long long w1  = (long long)VT_BASE % mod;            // pow(VT,1,mod)
    long long h;
    if (g < 4) {                                               // bigram
        h = (t0 + t1 * w1) % mod;
    } else {                                                   // trigram
        const long long w2 = (w1 * w1) % mod;                  // pow(VT,2,mod)
        h = (t0 + t1 * w1 + t2 * w2) % mod;
    }
    const long long row = h + (long long)g * M_BASE + (long long)g * g; // + offset

    const float* src = oe_weight + row * OH_DIM;
    unsigned short* dst = A + ((size_t)g * S_LEN + s) * OH_DIM;

    const int t = threadIdx.x;                                 // 0..255 -> 2 elems
    float2 v = *(const float2*)(src + t * 2);
    unsigned int packed = (unsigned int)f2bf(v.x) | ((unsigned int)f2bf(v.y) << 16);
    *(unsigned int*)(dst + t * 2) = packed;
}

// ---------------------------------------------------------------------------
// Kernel 2: Bt[g][n][k] = bf16(oe_projection[g][k][n]) — LDS-tiled transpose.
// ---------------------------------------------------------------------------
__global__ __launch_bounds__(256)
void transpose_convert_B(const float* __restrict__ proj,
                         unsigned short* __restrict__ Bt)
{
    __shared__ float tile[32][33];
    const int g  = blockIdx.z;
    const int n0 = blockIdx.x * 32;
    const int k0 = blockIdx.y * 32;
    const int tx = threadIdx.x;          // 0..31
    const int ty = threadIdx.y;          // 0..7

    const float* src = proj + (size_t)g * OH_DIM * D_DIM;
#pragma unroll
    for (int i = 0; i < 32; i += 8)
        tile[ty + i][tx] = src[(size_t)(k0 + ty + i) * D_DIM + n0 + tx];
    __syncthreads();

    unsigned short* dst = Bt + (size_t)g * D_DIM * OH_DIM;
#pragma unroll
    for (int i = 0; i < 32; i += 8)
        dst[(size_t)(n0 + ty + i) * OH_DIM + k0 + tx] = f2bf(tile[tx][ty + i]);
}

// ---------------------------------------------------------------------------
// Kernel 3: bf16 WMMA GEMM, software-pipelined:
//   global loads (step i+1) -> [12 ds_load frags | 8 WMMA] -> ds_store (i+1)
//   -> barrier.  Double-buffered LDS, fused word gather + 1/9 scale epilogue.
// ---------------------------------------------------------------------------
__global__ __launch_bounds__(256)
void oe_gemm(const unsigned short* __restrict__ A,   // [G][S][OH] bf16
             const unsigned short* __restrict__ Bt,  // [G][D][OH] bf16
             const int*            __restrict__ input_ids,
             const float*          __restrict__ word_weight,
             float*                __restrict__ out)
{
    constexpr int LD     = 40;                   // ushort stride: 80B, 16B-aligned
    constexpr int NSTEPS = G_GRAMS * (OH_DIM / 32);   // 128 k-steps total
    __shared__ unsigned short As[2][128 * LD];   // 2 x 10 KB
    __shared__ unsigned short Bs[2][128 * LD];   // 2 x 10 KB

    const int tid = threadIdx.x;
    const int w   = tid >> 5;                    // wave id 0..7
    const int l   = tid & 31;                    // lane
    const int wr  = w & 3;                       // wave row 0..3  (32 rows each)
    const int wc  = w >> 2;                      // wave col 0..1  (64 cols each)
    const bool hi = (l >= 16);
    const int  m  = l & 15;

    const int rowBase = blockIdx.y * 128;        // seq dim
    const int colBase = blockIdx.x * 128;        // D dim

    // CDNA5 bf16 WMMA fragment K offsets
    const int ac0 = hi ? 16 : 0;                 // A: K chunk 1 (K 0-7 / 16-23)
    const int ac1 = hi ? 24 : 8;                 // A: K chunk 2 (K 8-15 / 24-31)
    const int bk0 = hi ? 16 : 0;                 // B: contiguous 16 K

    // per-thread staging coordinates (two 16B chunks per array per step)
    const int srow0 = tid >> 2;                  // rows 0..63
    const int srow1 = srow0 + 64;                // rows 64..127
    const int sko   = (tid & 3) * 8;

    const int arow0 = wr * 32 + m;               // A frag rows (r=0 / r=1)
    const int arow1 = arow0 + 16;
    const int bcol  = wc * 64 + m;               // B frag col base

    v8f acc[2][4];
#pragma unroll
    for (int r = 0; r < 2; ++r)
#pragma unroll
        for (int c = 0; c < 4; ++c)
            acc[r][c] = (v8f)0.0f;

    union Frag { v16bf v; us8 h[2]; };

    us8 sa0, sa1, sb0, sb1;                      // staging registers

    // phase 1: issue global loads for k-step `idx` (no waiting here)
    auto gload = [&](int idx) {
        const int g  = idx >> 4;                 // 16 k-steps per gram
        const int kt = (idx & 15) * 32;
        const unsigned short* Ag = A  + ((size_t)g * S_LEN + rowBase) * OH_DIM + kt;
        const unsigned short* Bg = Bt + ((size_t)g * D_DIM + colBase) * OH_DIM + kt;
        sa0 = *(const us8*)&Ag[(size_t)srow0 * OH_DIM + sko];
        sa1 = *(const us8*)&Ag[(size_t)srow1 * OH_DIM + sko];
        sb0 = *(const us8*)&Bg[(size_t)srow0 * OH_DIM + sko];
        sb1 = *(const us8*)&Bg[(size_t)srow1 * OH_DIM + sko];
        if ((idx & 15) != 15) {                  // prefetch next k-step into L2
            __builtin_prefetch(&Ag[(size_t)(tid >> 1) * OH_DIM + 32], 0, 0);
            __builtin_prefetch(&Bg[(size_t)(tid >> 1) * OH_DIM + 32], 0, 0);
        }
    };
    // phase 2: commit staged registers to LDS buffer `buf`
    auto sstore = [&](int buf) {
        *(us8*)&As[buf][srow0 * LD + sko] = sa0;
        *(us8*)&As[buf][srow1 * LD + sko] = sa1;
        *(us8*)&Bs[buf][srow0 * LD + sko] = sb0;
        *(us8*)&Bs[buf][srow1 * LD + sko] = sb1;
    };

    gload(0);
    sstore(0);
    __syncthreads();

    int buf = 0;
    for (int i = 0; i < NSTEPS; ++i) {
        const bool more = (i + 1 < NSTEPS);
        if (more)
            gload(i + 1);                        // global loads in flight over WMMAs

        // ---- all 12 fragment ds_loads issued before any WMMA ----
        Frag a0, a1, b0, b1, b2, b3;
        a0.h[0] = *(const us8*)&As[buf][arow0 * LD + ac0];
        a0.h[1] = *(const us8*)&As[buf][arow0 * LD + ac1];
        a1.h[0] = *(const us8*)&As[buf][arow1 * LD + ac0];
        a1.h[1] = *(const us8*)&As[buf][arow1 * LD + ac1];
        b0.h[0] = *(const us8*)&Bs[buf][(bcol +  0) * LD + bk0];
        b0.h[1] = *(const us8*)&Bs[buf][(bcol +  0) * LD + bk0 + 8];
        b1.h[0] = *(const us8*)&Bs[buf][(bcol + 16) * LD + bk0];
        b1.h[1] = *(const us8*)&Bs[buf][(bcol + 16) * LD + bk0 + 8];
        b2.h[0] = *(const us8*)&Bs[buf][(bcol + 32) * LD + bk0];
        b2.h[1] = *(const us8*)&Bs[buf][(bcol + 32) * LD + bk0 + 8];
        b3.h[0] = *(const us8*)&Bs[buf][(bcol + 48) * LD + bk0];
        b3.h[1] = *(const us8*)&Bs[buf][(bcol + 48) * LD + bk0 + 8];
        __builtin_amdgcn_sched_barrier(0);       // keep loads ahead of WMMAs

        // consume in load order -> progressive s_wait_dscnt counts
        acc[0][0] = __builtin_amdgcn_wmma_f32_16x16x32_bf16(false, a0.v, false, b0.v, (short)0, acc[0][0], false, false);
        acc[1][0] = __builtin_amdgcn_wmma_f32_16x16x32_bf16(false, a1.v, false, b0.v, (short)0, acc[1][0], false, false);
        acc[0][1] = __builtin_amdgcn_wmma_f32_16x16x32_bf16(false, a0.v, false, b1.v, (short)0, acc[0][1], false, false);
        acc[1][1] = __builtin_amdgcn_wmma_f32_16x16x32_bf16(false, a1.v, false, b1.v, (short)0, acc[1][1], false, false);
        acc[0][2] = __builtin_amdgcn_wmma_f32_16x16x32_bf16(false, a0.v, false, b2.v, (short)0, acc[0][2], false, false);
        acc[1][2] = __builtin_amdgcn_wmma_f32_16x16x32_bf16(false, a1.v, false, b2.v, (short)0, acc[1][2], false, false);
        acc[0][3] = __builtin_amdgcn_wmma_f32_16x16x32_bf16(false, a0.v, false, b3.v, (short)0, acc[0][3], false, false);
        acc[1][3] = __builtin_amdgcn_wmma_f32_16x16x32_bf16(false, a1.v, false, b3.v, (short)0, acc[1][3], false, false);
        __builtin_amdgcn_sched_barrier(0);       // ds_store stays after WMMAs

        if (more)
            sstore(buf ^ 1);                     // loadcnt wait lands *after* WMMAs
        __syncthreads();                         // one barrier per step
        buf ^= 1;
    }

    // ---- epilogue: fused word gather + mean-of-9 scale ----
    const int mbase = hi ? 8 : 0;                // C/D layout: lanes>=16 hold M=v+8
#pragma unroll
    for (int r = 0; r < 2; ++r) {
        const int srow = rowBase + wr * 32 + r * 16 + mbase;
        size_t wrow[8];
#pragma unroll
        for (int v = 0; v < 8; ++v)
            wrow[v] = (size_t)input_ids[srow + v] * D_DIM;
#pragma unroll
        for (int c = 0; c < 4; ++c) {
            const int col = colBase + wc * 64 + c * 16 + m;
#pragma unroll
            for (int v = 0; v < 8; ++v) {
                const float wv = word_weight[wrow[v] + col];
                out[(size_t)(srow + v) * D_DIM + col] =
                    (wv + acc[r][c][v]) * (1.0f / 9.0f);
            }
        }
    }
}

// ---------------------------------------------------------------------------
extern "C" void kernel_launch(void* const* d_in, const int* in_sizes, int n_in,
                              void* d_out, int out_size, void* d_ws, size_t ws_size,
                              hipStream_t stream)
{
    const int*   input_ids   = (const int*)  d_in[0];   // [4096]
    const float* word_weight = (const float*)d_in[1];   // [32000, 4096]
    const float* oe_weight   = (const float*)d_in[2];   // [400232, 512]
    const float* oe_proj     = (const float*)d_in[3];   // [8, 512, 4096]
    float*       out         = (float*)d_out;           // [4096, 4096]

    // workspace: A_bf16 [8][4096][512] (32 MB) | Bt_bf16 [8][4096][512] (32 MB)
    const size_t A_BYTES = (size_t)G_GRAMS * S_LEN * OH_DIM * sizeof(unsigned short);
    unsigned short* A  = (unsigned short*)d_ws;
    unsigned short* Bt = (unsigned short*)((char*)d_ws + A_BYTES);

    gather_convert_A<<<dim3(S_LEN, G_GRAMS), 256, 0, stream>>>(input_ids, oe_weight, A);

    transpose_convert_B<<<dim3(D_DIM / 32, OH_DIM / 32, G_GRAMS), dim3(32, 8), 0, stream>>>(
        oe_proj, Bt);

    oe_gemm<<<dim3(D_DIM / 128, S_LEN / 128), 256, 0, stream>>>(
        A, Bt, input_ids, word_weight, out);
}
